// AdvancedFastMQA_61383672594957
// MI455X (gfx1250) — matrix operand order repaired and checked by reading, not compiled, not run on hardware
//
#include <hip/hip_runtime.h>
#include <cstdint>
#include <cstddef>

typedef __attribute__((ext_vector_type(16))) _Float16     v16h;
typedef __attribute__((ext_vector_type(8)))  float        v8f;
typedef __attribute__((ext_vector_type(4)))  unsigned int v4u;
typedef __attribute__((ext_vector_type(8)))  int          v8i;
typedef __attribute__((ext_vector_type(4)))  int          v4i;

#define B_   8
#define S_   1024
#define H_   32
#define DH_  128
#define HD_  4096
#define SCALE_ 0.08838834764831845f   // 1/sqrt(128)

#if defined(__HIP_DEVICE_COMPILE__) && __has_builtin(__builtin_amdgcn_tensor_load_to_lds)
#define USE_TDM 1
#else
#define USE_TDM 0
#endif

// ---- fragment helpers ------------------------------------------------------
// A-matrix 16x32 f16 layout (ISA 7.12.2): lane<16 -> row=lane, K = [0..7] U [16..23];
// lane>=16 -> row=lane-16, K = [8..15] U [24..31].  Two 16B chunks 16 elems apart.
// B-matrix 32x16 f16 layout: lane holds column n=lane&15; lanes<16 K = [0..7] U [16..23],
// lanes>=16 K = [8..15] U [24..31].  Two 16B chunks 16 elems apart.
static __device__ __forceinline__ v16h ldpair(const _Float16* p0, const _Float16* p1) {
  union { uint4 q[2]; v16h v; } u;
  u.q[0] = *(const uint4*)p0;
  u.q[1] = *(const uint4*)p1;
  return u.v;
}
static __device__ __forceinline__ v8f vzero8() {
  v8f z = {0.f, 0.f, 0.f, 0.f, 0.f, 0.f, 0.f, 0.f};
  return z;
}

// ---- f32 -> f16 conversion -------------------------------------------------
__global__ void cvt_f16_kernel(const float* __restrict__ s, _Float16* __restrict__ d, size_t n) {
  size_t t = (size_t)blockIdx.x * blockDim.x + threadIdx.x;
  if (t < n) d[t] = (_Float16)s[t];
}

// ---- tiled WMMA GEMM: C[M,N](f32) = A[M,K](f16) * Bw[N,K](f16)^T -----------
#define BM 128
#define BN 128
#define BK 64
#define LSTR 72   // 64 + 8 pad: row stride 144B -> bank stride 36 -> conflict-free b128 frags

#if USE_TDM
// TDM 2D tile load: BM rows x BK f16 cols from row-major (stride k_stride elems)
// into LDS with 4-DWORD pad after every 32 DWORDs -> LDS row stride = LSTR f16.
// D# layout per CDNA5 ISA.
__device__ __forceinline__ void tdm_load_2d(const _Float16* g, void* l, unsigned k_stride) {
  unsigned long long ga = (unsigned long long)(size_t)g;
  v4u g0;
  g0.x = 1u;                                            // count=1, user descriptor
  g0.y = (unsigned)(size_t)l;                           // lds_addr (bytes)
  g0.z = (unsigned)ga;                                  // global_addr[31:0]
  g0.w = (unsigned)((ga >> 32) & 0x01FFFFFFu) | (2u << 30);  // addr[56:32], type=2
  const unsigned td0 = 0x40000000u, td1 = 0x40000000u;  // huge dims: tile fully in-bounds
  v8i g1;
  g1[0] = (int)((1u << 16)      // data_size = 2 bytes
              | (1u << 20)      // pad_enable
              | (4u << 22)      // pad_interval: every 32 DWORDs (one 64-f16 row)
              | (3u << 25));    // pad_amount: 4 DWORDs (8 f16)
  g1[1] = (int)((td0 & 0xFFFFu) << 16);                       // tensor_dim0[15:0]
  g1[2] = (int)((td0 >> 16) | ((td1 & 0xFFFFu) << 16));       // td0.hi | td1.lo
  g1[3] = (int)((td1 >> 16) | ((unsigned)BK << 16));          // td1.hi | tile_dim0=64
  g1[4] = (int)(unsigned)BM;                                  // tile_dim1=128, tile_dim2=0
  g1[5] = (int)k_stride;                                      // tensor_dim0_stride[31:0]
  g1[6] = 0;
  g1[7] = 0;
  v4i z4 = {0, 0, 0, 0};
#if __clang_major__ >= 23
  v8i z8 = {0, 0, 0, 0, 0, 0, 0, 0};
  __builtin_amdgcn_tensor_load_to_lds(g0, g1, z4, z4, z8, 0);
#else
  __builtin_amdgcn_tensor_load_to_lds(g0, g1, z4, z4, 0);
#endif
}
#endif

__device__ __forceinline__ void load_tile(const _Float16* __restrict__ A,
                                          const _Float16* __restrict__ Bw,
                                          _Float16 (*As)[LSTR], _Float16 (*Bs)[LSTR],
                                          int cm, int cn, int K, int k0, int tid, int pf) {
  int r = tid >> 3;
  int c = (tid & 7) << 3;
#pragma unroll
  for (int it = 0; it < 4; ++it) {
    int row = r + it * 32;
    const _Float16* ap = A  + (size_t)(cm + row) * K + k0 + c;
    const _Float16* bp = Bw + (size_t)(cn + row) * K + k0 + c;
    *(uint4*)&As[row][c] = *(const uint4*)ap;
    *(uint4*)&Bs[row][c] = *(const uint4*)bp;
    if (pf) {                       // global_prefetch_b8: pull tile kt+2 toward L2
      __builtin_prefetch(ap + BK, 0, 1);
      __builtin_prefetch(bp + BK, 0, 1);
    }
  }
}

__global__ __launch_bounds__(256) void gemm_f16_kernel(const _Float16* __restrict__ A,
                                                       const _Float16* __restrict__ Bw,
                                                       float* __restrict__ C,
                                                       int M, int N, int K) {
  __shared__ _Float16 As[2][BM][LSTR];
  __shared__ _Float16 Bs[2][BM][LSTR];
  int tid  = threadIdx.x;
  int lane = tid & 31, wid = tid >> 5;
  int wm = wid >> 1, wn = wid & 1;          // 4x2 wave grid -> wave tile 32(M) x 64(N)
  int cm = blockIdx.y * BM, cn = blockIdx.x * BN;
  int ar   = lane & 15;
  int ka   = (lane >> 4) << 3;              // A frag K base per lane-half
  int kb16 = (lane >> 4) << 3;              // B frag K base per lane-half (table order, like A)

  v8f acc[2][4];
#pragma unroll
  for (int i = 0; i < 2; ++i)
#pragma unroll
    for (int j = 0; j < 4; ++j) acc[i][j] = vzero8();

  int nt = K / BK;
#if USE_TDM
  bool issuer = (wid == 0);                 // one wave drives the Tensor Data Mover
  if (issuer) {
    tdm_load_2d(A  + (size_t)cm * K, &As[0][0][0], (unsigned)K);
    tdm_load_2d(Bw + (size_t)cn * K, &Bs[0][0][0], (unsigned)K);
    __builtin_amdgcn_s_wait_tensorcnt(0);
  }
  __syncthreads();
#else
  load_tile(A, Bw, As[0], Bs[0], cm, cn, K, 0, tid, nt > 1);
  __syncthreads();
#endif

  for (int kt = 0; kt < nt; ++kt) {
    int cur = kt & 1;
#if USE_TDM
    if (kt + 1 < nt && issuer) {            // DMA tile kt+1 while computing tile kt
      tdm_load_2d(A  + (size_t)cm * K + (kt + 1) * BK, &As[cur ^ 1][0][0], (unsigned)K);
      tdm_load_2d(Bw + (size_t)cn * K + (kt + 1) * BK, &Bs[cur ^ 1][0][0], (unsigned)K);
    }
#else
    if (kt + 1 < nt)
      load_tile(A, Bw, As[cur ^ 1], Bs[cur ^ 1], cm, cn, K, (kt + 1) * BK, tid, kt + 2 < nt);
#endif
#pragma unroll
    for (int ks = 0; ks < BK; ks += 32) {
      v16h af[2], bf[4];
#pragma unroll
      for (int i = 0; i < 2; ++i) {
        const _Float16* p = &As[cur][wm * 32 + i * 16 + ar][ks + ka];
        af[i] = ldpair(p, p + 16);
      }
#pragma unroll
      for (int j = 0; j < 4; ++j) {
        const _Float16* p = &Bs[cur][wn * 64 + j * 16 + ar][ks + kb16];
        bf[j] = ldpair(p, p + 16);
      }
#pragma unroll
      for (int i = 0; i < 2; ++i)
#pragma unroll
        for (int j = 0; j < 4; ++j)
          acc[i][j] = __builtin_amdgcn_wmma_f32_16x16x32_f16(
              false, af[i], false, bf[j], (short)0, acc[i][j], false, false);
    }
#if USE_TDM
    if (kt + 1 < nt && issuer) __builtin_amdgcn_s_wait_tensorcnt(0);
#endif
    __syncthreads();
  }

  // C/D layout: lane&15 = N, VGPR r -> M = r (+8 for upper lane half)
  int mh = (lane >> 4) << 3;
#pragma unroll
  for (int i = 0; i < 2; ++i)
#pragma unroll
    for (int j = 0; j < 4; ++j) {
      int n  = cn + wn * 64 + j * 16 + ar;
      int m0 = cm + wm * 32 + i * 16 + mh;
#pragma unroll
      for (int r = 0; r < 8; ++r)
        C[(size_t)(m0 + r) * N + n] = acc[i][j][r];
    }
}

// ---- RoPE + layout transform + f16 conversion ------------------------------
// Qf [B,S,H*DH] -> Qr [B,H,S,DH] f16 (roped)
// Kf [B,S,DH]   -> Kr [B,S,DH]   f16 (roped)
// Vf [B,S,DH]   -> Vt [B,DH,S]   f16 (transposed so attention B-frags are contiguous)
__global__ void rope_cvt_kernel(const float* __restrict__ Qf, const float* __restrict__ Kf,
                                const float* __restrict__ Vf, _Float16* __restrict__ Qr,
                                _Float16* __restrict__ Kr, _Float16* __restrict__ Vt) {
  int t = blockIdx.x * 256 + threadIdx.x;      // B*S*DH threads
  int d = t & 127;
  int s = (t >> 7) & 1023;
  int b = t >> 17;
  float ang = (float)s * __powf(10000.0f, -(float)(2 * (d >> 1)) * (1.0f / 128.0f));
  float sn, cs;
  __sincosf(ang, &sn, &cs);
  int   pr  = (d < 64) ? d + 64 : d - 64;
  float sgn = (d < 64) ? -1.0f : 1.0f;

  size_t kv = ((size_t)b * S_ + s) * DH_;
  float kq = Kf[kv + d], kp = Kf[kv + pr];
  Kr[kv + d] = (_Float16)(kq * cs + sgn * kp * sn);
  Vt[((size_t)b * DH_ + d) * S_ + s] = (_Float16)Vf[kv + d];

  size_t qrow = ((size_t)b * S_ + s) * HD_;
#pragma unroll 4
  for (int h = 0; h < H_; ++h) {
    float q  = Qf[qrow + h * DH_ + d];
    float qp = Qf[qrow + h * DH_ + pr];
    Qr[(((size_t)(b * H_ + h)) * S_ + s) * DH_ + d] = (_Float16)(q * cs + sgn * qp * sn);
  }
}

// ---- one sliding window: wout = rownorm(sigmoid(scale*Q K^T)) @ V ----------
// 4 waves/block, each wave owns 16 query rows; block covers 64 rows of the window.
__global__ __launch_bounds__(128) void attn_window_kernel(
    const _Float16* __restrict__ Qr, const _Float16* __restrict__ Kr,
    const _Float16* __restrict__ Vt, float* __restrict__ Wout, int i0, int L) {
  __shared__ _Float16 P[64][520];      // sigmoid probs (unnormalized), padded rows
  __shared__ float rs2[64][2];
  __shared__ float rs[64];

  int b = blockIdx.z, h = blockIdx.y;
  int lane = threadIdx.x & 31, wid = threadIdx.x >> 5;
  int q0 = blockIdx.x * 64 + wid * 16;          // q row base within window
  int ar = lane & 15;
  int ka = (lane >> 4) << 3;
  int kb16 = (lane >> 4) << 3;
  int mh = ka;

  // Q fragments for this wave's 16 rows, full DH=128 (4 k-chunks), kept in VGPRs
  const _Float16* Qbase = Qr + (((size_t)(b * H_ + h)) * S_ + i0 + q0) * DH_;
  v16h aq[4];
#pragma unroll
  for (int kc = 0; kc < 4; ++kc) {
    const _Float16* p = Qbase + (size_t)ar * DH_ + kc * 32 + ka;
    aq[kc] = ldpair(p, p + 16);
  }

  // pass 1: scores -> sigmoid -> LDS
  const _Float16* Kbase = Kr + ((size_t)b * S_ + i0) * DH_;
  for (int nb = 0; nb < (L >> 4); ++nb) {
    v8f acc = vzero8();
    const _Float16* kp = Kbase + (size_t)(nb * 16 + ar) * DH_ + kb16;
#pragma unroll
    for (int kc = 0; kc < 4; ++kc) {
      v16h bk = ldpair(kp + kc * 32, kp + kc * 32 + 16);
      acc = __builtin_amdgcn_wmma_f32_16x16x32_f16(false, aq[kc], false, bk,
                                                   (short)0, acc, false, false);
    }
    int n = nb * 16 + ar;
#pragma unroll
    for (int r = 0; r < 8; ++r) {
      float p = 1.0f / (1.0f + __expf(-acc[r] * SCALE_));
      P[wid * 16 + mh + r][n] = (_Float16)p;
    }
  }
  __syncthreads();

  // row sums (normalization is linear -> divide after the P@V GEMM)
  {
    int row = threadIdx.x >> 1, hs = threadIdx.x & 1, hl = L >> 1;
    float sum = 0.0f;
    for (int k = hs * hl; k < hs * hl + hl; ++k) sum += (float)P[row][k];
    rs2[row][hs] = sum;
  }
  __syncthreads();
  if (threadIdx.x < 64) rs[threadIdx.x] = rs2[threadIdx.x][0] + rs2[threadIdx.x][1];
  __syncthreads();

  float inv[8];
#pragma unroll
  for (int r = 0; r < 8; ++r) inv[r] = 1.0f / rs[wid * 16 + mh + r];

  // pass 2: wout = P @ V  (A frags from LDS, B frags contiguous from V^T)
  const _Float16* Vbase = Vt + (size_t)b * DH_ * S_ + i0;
  float* Wbase = Wout + (((size_t)(b * H_ + h)) * 512 + q0) * DH_;
#pragma unroll 2
  for (int nt2 = 0; nt2 < 8; ++nt2) {
    v8f acc = vzero8();
    int dcol = nt2 * 16 + ar;
    const _Float16* vp = Vbase + (size_t)dcol * S_ + kb16;
    const _Float16* pp = &P[wid * 16 + ar][ka];
    for (int kc = 0; kc < (L >> 5); ++kc) {
      v16h ap = ldpair(pp + kc * 32, pp + kc * 32 + 16);
      v16h bv = ldpair(vp + kc * 32, vp + kc * 32 + 16);
      acc = __builtin_amdgcn_wmma_f32_16x16x32_f16(false, ap, false, bv,
                                                   (short)0, acc, false, false);
    }
#pragma unroll
    for (int r = 0; r < 8; ++r)
      Wbase[(size_t)(mh + r) * DH_ + dcol] = acc[r] * inv[r];
  }
}

// ---- blend overlapping windows, emit f16 [B,S,H,DH] for O-proj -------------
__global__ void blend_cvt_kernel(const float* __restrict__ W0, const float* __restrict__ W1,
                                 const float* __restrict__ W2, const float* __restrict__ W3,
                                 _Float16* __restrict__ attnb) {
  size_t t = (size_t)blockIdx.x * 256 + threadIdx.x;   // B*S*H*DH
  int d = (int)(t & 127);
  int h = (int)(t >> 7) & 31;
  int s = (int)(t >> 12) & 1023;
  int b = (int)(t >> 22);
  int bI = s >> 8;
  size_t base = ((size_t)(b * H_ + h)) * 512;
  float val;
  if (bI == 0) {
    val = W0[(base + s) * DH_ + d];
  } else {
    const float* Wp = (bI == 1) ? W0 : (bI == 2) ? W1 : W2;
    const float* Wc = (bI == 1) ? W1 : (bI == 2) ? W2 : W3;
    float a = (float)(s & 255) * (1.0f / 255.0f);
    int prow = s - 256 * (bI - 1);   // row in previous window (256..511)
    int crow = s - 256 * bI;         // row in current window  (0..255)
    val = (1.0f - a) * Wp[(base + prow) * DH_ + d] + a * Wc[(base + crow) * DH_ + d];
  }
  attnb[t] = (_Float16)val;
}

// ---- driver ----------------------------------------------------------------
extern "C" void kernel_launch(void* const* d_in, const int* in_sizes, int n_in,
                              void* d_out, int out_size, void* d_ws, size_t ws_size,
                              hipStream_t stream) {
  const float* x  = (const float*)d_in[0];
  const float* Wq = (const float*)d_in[1];
  const float* Wk = (const float*)d_in[2];
  const float* Wv = (const float*)d_in[3];
  const float* Wo = (const float*)d_in[4];
  float* out = (float*)d_out;

  char* ws = (char*)d_ws;
  size_t off = 0;
  auto alloc = [&](size_t bytes) -> void* {
    void* p = ws + off;
    off += (bytes + 255) & ~(size_t)255;
    return p;
  };
  const size_t NX  = (size_t)B_ * S_ * HD_;   // 33,554,432
  const size_t NKV = (size_t)B_ * S_ * DH_;   // 1,048,576
  const size_t NWB = (size_t)B_ * H_ * 512 * DH_;

  _Float16* xb   = (_Float16*)alloc(NX * 2);
  _Float16* Wqb  = (_Float16*)alloc((size_t)HD_ * HD_ * 2);
  _Float16* Wkb  = (_Float16*)alloc((size_t)DH_ * HD_ * 2);
  _Float16* Wvb  = (_Float16*)alloc((size_t)DH_ * HD_ * 2);
  _Float16* Wob  = (_Float16*)alloc((size_t)HD_ * HD_ * 2);
  float*    Qf   = (float*)alloc(NX * 4);
  float*    Kf   = (float*)alloc(NKV * 4);
  float*    Vf   = (float*)alloc(NKV * 4);
  _Float16* Qrb  = (_Float16*)alloc(NX * 2);
  _Float16* Krb  = (_Float16*)alloc(NKV * 2);
  _Float16* Vtb  = (_Float16*)alloc(NKV * 2);
  float*    W0   = (float*)alloc(NWB * 4);
  float*    W1   = (float*)alloc(NWB * 4);
  float*    W2   = (float*)alloc(NWB * 4);
  float*    W3   = (float*)alloc(NWB * 4);
  _Float16* attnb = (_Float16*)alloc(NX * 2);

  auto cvt = [&](const float* s, _Float16* dst, size_t n) {
    cvt_f16_kernel<<<dim3((unsigned)((n + 255) / 256)), dim3(256), 0, stream>>>(s, dst, n);
  };
  cvt(x,  xb,  NX);
  cvt(Wq, Wqb, (size_t)HD_ * HD_);
  cvt(Wk, Wkb, (size_t)DH_ * HD_);
  cvt(Wv, Wvb, (size_t)DH_ * HD_);
  cvt(Wo, Wob, (size_t)HD_ * HD_);

  const int M = B_ * S_;
  gemm_f16_kernel<<<dim3(HD_ / BN, M / BM), 256, 0, stream>>>(xb, Wqb, Qf, M, HD_, HD_);
  gemm_f16_kernel<<<dim3(1,        M / BM), 256, 0, stream>>>(xb, Wkb, Kf, M, DH_, HD_);
  gemm_f16_kernel<<<dim3(1,        M / BM), 256, 0, stream>>>(xb, Wvb, Vf, M, DH_, HD_);

  rope_cvt_kernel<<<dim3((unsigned)(NKV / 256)), 256, 0, stream>>>(Qf, Kf, Vf, Qrb, Krb, Vtb);

  attn_window_kernel<<<dim3(8, H_, B_), 128, 0, stream>>>(Qrb, Krb, Vtb, W0,   0, 512);
  attn_window_kernel<<<dim3(8, H_, B_), 128, 0, stream>>>(Qrb, Krb, Vtb, W1, 256, 512);
  attn_window_kernel<<<dim3(8, H_, B_), 128, 0, stream>>>(Qrb, Krb, Vtb, W2, 512, 512);
  attn_window_kernel<<<dim3(4, H_, B_), 128, 0, stream>>>(Qrb, Krb, Vtb, W3, 768, 256);

  blend_cvt_kernel<<<dim3((unsigned)(NX / 256)), 256, 0, stream>>>(W0, W1, W2, W3, attnb);

  gemm_f16_kernel<<<dim3(HD_ / BN, M / BM), 256, 0, stream>>>(attnb, Wob, out, M, HD_, HD_);

  (void)in_sizes; (void)n_in; (void)out_size; (void)ws_size;
}
